// FusionFeedForward_49941879718222
// MI455X (gfx1250) — matrix operand order repaired and checked by reading, not compiled, not run on hardware
//
#include <hip/hip_runtime.h>
#include <hip/hip_bf16.h>
#include <stdint.h>

// ---------------- problem geometry (FuseFormer) ----------------
#define D_MODEL 512
#define HD      1960
#define HDP     1984     // HD padded to a multiple of 32 (zero-filled) for GEMM2 K-loop
#define KK      7
#define SS      3
#define PP      3
#define HH      60
#define WW      108
#define NHH     20
#define NWW     36
#define NVEC    720      // 20*36
#define CCH     40       // HD / 49
#define B2N     20       // 4 * 3600 / 720
#define MROWS   14400    // 4 * 3600

// ---------------- WMMA types ----------------
typedef __attribute__((ext_vector_type(16))) __bf16 v16bf;
typedef __attribute__((ext_vector_type(8)))  float  v8f;
typedef int vi4 __attribute__((__vector_size__(16)));   // matches builtin param type

union FragU {
  uint4 q[2];
  v16bf v;
};

#define AS1 __attribute__((address_space(1)))
#define AS3 __attribute__((address_space(3)))

#if __has_builtin(__builtin_amdgcn_global_load_async_to_lds_b128) && \
    __has_builtin(__builtin_amdgcn_s_wait_asynccnt)
#define USE_ASYNC_LDS 1
#define ASYNC_CP128(gsrc, ldst) \
  __builtin_amdgcn_global_load_async_to_lds_b128((AS1 vi4*)(gsrc), (AS3 vi4*)(ldst), 0, 0)
#else
#define USE_ASYNC_LDS 0
#endif

__device__ __forceinline__ unsigned short f2bf(float f) {
  unsigned int u = __builtin_bit_cast(unsigned int, f);
  unsigned int r = 0x7FFFu + ((u >> 16) & 1u);   // round-to-nearest-even
  return (unsigned short)((u + r) >> 16);
}

// ---------------- elementwise f32 -> bf16 ----------------
__global__ void k_f32_to_bf16(const float* __restrict__ in,
                              unsigned short* __restrict__ out, size_t n) {
  size_t i = (size_t)blockIdx.x * blockDim.x + threadIdx.x;
  if (i < n) out[i] = f2bf(in[i]);
}

// in: [K][N] f32 row-major -> out: [N][Kpad] bf16 (transposed, zero pad k>=K)
__global__ void k_transpose_to_bf16(const float* __restrict__ in,
                                    unsigned short* __restrict__ out,
                                    int K, int N, int Kpad) {
  size_t i = (size_t)blockIdx.x * blockDim.x + threadIdx.x;
  size_t total = (size_t)N * Kpad;
  if (i >= total) return;
  int k = (int)(i % Kpad);
  int n = (int)(i / Kpad);
  out[i] = (k < K) ? f2bf(in[(size_t)k * N + n]) : (unsigned short)0;
}

// ---------------- bf16 WMMA GEMM: D = A * Bt^T + bias ----------------
// A : [M][K] bf16, Bt : [N][K] bf16, D : [M][N] f32.  K % 32 == 0.
#define TILE_M 128
#define TILE_N 128
#define TILE_K 32
#define LDSS   40   // bf16 elems per LDS row (32 data + 8 pad): 80B rows, 16B-aligned, bank-spread
#define NBUF   4    // 4-deep LDS pipeline: 1 barrier per K-step, prefetch depth 2

__global__ __launch_bounds__(256)
void k_gemm_bf16(const unsigned short* __restrict__ A,
                 const unsigned short* __restrict__ Bt,
                 const float* __restrict__ bias,
                 float* __restrict__ D,
                 int M, int N, int K) {
  __shared__ alignas(16) unsigned short lA[NBUF][TILE_M * LDSS];
  __shared__ alignas(16) unsigned short lB[NBUF][TILE_N * LDSS];

  const int tid  = threadIdx.x;
  const int lane = tid & 31;
  const int wave = tid >> 5;
  const int wm = wave & 3;     // 4 M-slots of 32 rows
  const int wn = wave >> 2;    // 2 N-slots of 64 cols

  const int nTilesN = (N + TILE_N - 1) / TILE_N;
  const int m0 = (blockIdx.x / nTilesN) * TILE_M;
  const int n0 = (blockIdx.x % nTilesN) * TILE_N;

  // staging coords: 256 threads cover 128 rows x 32 k (2 rows per thread, 8 bf16 per load)
  const int sRow0 = tid >> 2;              // 0..63
  const int sRow1 = sRow0 + 64;            // 64..127
  const int sSeg  = (tid & 3) * 8;         // k element offset 0/8/16/24
  const int lOff0 = sRow0 * LDSS + sSeg;
  const int lOff1 = sRow1 * LDSS + sSeg;

  // clamp rows: loads always in-bounds; garbage rows/cols are never stored
  const int aRow0 = min(m0 + sRow0, M - 1);
  const int aRow1 = min(m0 + sRow1, M - 1);
  const int bRow0 = min(n0 + sRow0, N - 1);
  const int bRow1 = min(n0 + sRow1, N - 1);
  const unsigned short* aPtr0 = A  + (size_t)aRow0 * K + sSeg;
  const unsigned short* aPtr1 = A  + (size_t)aRow1 * K + sSeg;
  const unsigned short* bPtr0 = Bt + (size_t)bRow0 * K + sSeg;
  const unsigned short* bPtr1 = Bt + (size_t)bRow1 * K + sSeg;

  const int lrow  = lane & 15;
  const int khalf = (lane >> 4) << 3;   // A frag: 0 or 8
  const int koffB = (lane >> 4) << 3;   // B frag: 0 or 8

  v8f acc[2][4] = {};
  const int nSteps = K / TILE_K;

  auto computeTile = [&](int step) {
    const int cur = step & (NBUF - 1);
    // A fragments (ISA 16-bit A 16x32 layout): runs [khalf,+8) and [khalf+16,+8)
    FragU af[2];
    #pragma unroll
    for (int mt = 0; mt < 2; ++mt) {
      const int row = wm * 32 + mt * 16 + lrow;
      af[mt].q[0] = *(const uint4*)(&lA[cur][row * LDSS + khalf]);
      af[mt].q[1] = *(const uint4*)(&lA[cur][row * LDSS + khalf + 16]);
    }
    // B fragments (ISA B 32x16 layout): runs [koffB,+8) and [koffB+16,+8)
    FragU bfr[4];
    #pragma unroll
    for (int nt = 0; nt < 4; ++nt) {
      const int row = wn * 64 + nt * 16 + lrow;
      bfr[nt].q[0] = *(const uint4*)(&lB[cur][row * LDSS + koffB]);
      bfr[nt].q[1] = *(const uint4*)(&lB[cur][row * LDSS + koffB + 16]);
    }
    #pragma unroll
    for (int mt = 0; mt < 2; ++mt)
      #pragma unroll
      for (int nt = 0; nt < 4; ++nt)
        acc[mt][nt] = __builtin_amdgcn_wmma_f32_16x16x32_bf16(
            false, af[mt].v, false, bfr[nt].v, (short)0, acc[mt][nt], false, false);
  };

#if USE_ASYNC_LDS
  auto issueTile = [&](int step) {
    const int b = step & (NBUF - 1);
    const int k0 = step * TILE_K;
    ASYNC_CP128(aPtr0 + k0, &lA[b][lOff0]);
    ASYNC_CP128(aPtr1 + k0, &lA[b][lOff1]);
    ASYNC_CP128(bPtr0 + k0, &lB[b][lOff0]);
    ASYNC_CP128(bPtr1 + k0, &lB[b][lOff1]);
  };

  issueTile(0);
  if (nSteps > 1) issueTile(1);
  // steady state: branch-free, one barrier per K-step, two tiles in flight
  for (int i = 0; i < nSteps - 2; ++i) {
    issueTile(i + 2);
    __builtin_amdgcn_s_wait_asynccnt(8);   // tile i complete; i+1, i+2 may remain in flight
    __syncthreads();
    computeTile(i);
  }
  if (nSteps > 1) {
    __builtin_amdgcn_s_wait_asynccnt(4);   // tile nSteps-2 complete
    __syncthreads();
    computeTile(nSteps - 2);
  }
  __builtin_amdgcn_s_wait_asynccnt(0);     // final tile complete
  __syncthreads();
  computeTile(nSteps - 1);
#else
  // synchronous fallback: register-staged, 2 buffers, 2 barriers per step
  uint4 ra0 = *(const uint4*)(aPtr0);
  uint4 ra1 = *(const uint4*)(aPtr1);
  uint4 rb0 = *(const uint4*)(bPtr0);
  uint4 rb1 = *(const uint4*)(bPtr1);
  for (int i = 0; i < nSteps; ++i) {
    const int cur = i & 1;
    const int kNext = (i + 1) * TILE_K;
    *(uint4*)(&lA[cur][lOff0]) = ra0;
    *(uint4*)(&lA[cur][lOff1]) = ra1;
    *(uint4*)(&lB[cur][lOff0]) = rb0;
    *(uint4*)(&lB[cur][lOff1]) = rb1;
    if (i + 1 < nSteps) {
      ra0 = *(const uint4*)(aPtr0 + kNext);
      ra1 = *(const uint4*)(aPtr1 + kNext);
      rb0 = *(const uint4*)(bPtr0 + kNext);
      rb1 = *(const uint4*)(bPtr1 + kNext);
    }
    __syncthreads();
    computeTile(i);
    __syncthreads();
  }
#endif

  // epilogue: lane n = lrow; acc element r -> M = base + (lane<16 ? r : 8+r)
  const int rbase = (lane >> 4) << 3;
  #pragma unroll
  for (int nt = 0; nt < 4; ++nt) {
    const int gn = n0 + wn * 64 + nt * 16 + lrow;
    if (gn >= N) continue;
    const float bv = bias[gn];
    #pragma unroll
    for (int mt = 0; mt < 2; ++mt) {
      const int mbase = m0 + wm * 32 + mt * 16 + rbase;
      #pragma unroll
      for (int r = 0; r < 8; ++r) {
        const int gm = mbase + r;
        if (gm < M) D[(size_t)gm * N + gn] = acc[mt][nt][r] + bv;
      }
    }
  }
}

// ---------------- fold + normalize ----------------
// h: [B2][720][1960] f32 -> img: [B2][C][60][108] f32 (normalized)
__global__ void k_fold_norm(const float* __restrict__ h, float* __restrict__ img) {
  int idx = blockIdx.x * blockDim.x + threadIdx.x;
  const int total = B2N * CCH * HH * WW;
  if (idx >= total) return;
  int x = idx % WW;
  int t = idx / WW;
  int y = t % HH;  t /= HH;
  int c = t % CCH;
  int b2 = t / CCH;
  int yp = y + PP, xp = x + PP;
  float sum = 0.f;
  int cnt = 0;
  for (int ki = yp % SS; ki < KK; ki += SS) {
    int bi = (yp - ki) / SS;
    if (bi < 0 || bi >= NHH) continue;
    for (int kj = xp % SS; kj < KK; kj += SS) {
      int bj = (xp - kj) / SS;
      if (bj < 0 || bj >= NWW) continue;
      int nvec = bi * NWW + bj;
      int hd = c * (KK * KK) + ki * KK + kj;
      sum += h[((size_t)(b2 * NVEC + nvec)) * HD + hd];
      ++cnt;
    }
  }
  img[idx] = sum / (float)cnt;
}

// ---------------- unfold + relu -> bf16 (K padded to HDP, pad cols = 0) ----------------
__global__ void k_unfold_relu(const float* __restrict__ img,
                              unsigned short* __restrict__ h2) {
  size_t idx = (size_t)blockIdx.x * blockDim.x + threadIdx.x;
  const size_t total = (size_t)MROWS * HDP;
  if (idx >= total) return;
  int hd = (int)(idx % HDP);
  size_t row = idx / HDP;
  float v = 0.f;
  if (hd < HD) {
    int nvec = (int)(row % NVEC);
    int b2 = (int)(row / NVEC);
    int c = hd / (KK * KK), kk = hd % (KK * KK);
    int ki = kk / KK, kj = kk % KK;
    int bi = nvec / NWW, bj = nvec % NWW;
    int y = bi * SS + ki - PP;
    int x = bj * SS + kj - PP;
    if (y >= 0 && y < HH && x >= 0 && x < WW)
      v = img[(((size_t)b2 * CCH + c) * HH + y) * WW + x];
    v = v > 0.f ? v : 0.f;   // ReLU
  }
  h2[idx] = f2bf(v);
}

// ---------------- launcher ----------------
static inline size_t alignup(size_t x) { return (x + 255) & ~(size_t)255; }

extern "C" void kernel_launch(void* const* d_in, const int* in_sizes, int n_in,
                              void* d_out, int out_size, void* d_ws, size_t ws_size,
                              hipStream_t stream) {
  (void)in_sizes; (void)n_in; (void)out_size; (void)ws_size;
  const float* x  = (const float*)d_in[0];   // [14400][512]
  const float* W1 = (const float*)d_in[1];   // [512][1960]
  const float* b1 = (const float*)d_in[2];   // [1960]
  const float* W2 = (const float*)d_in[3];   // [1960][512]
  const float* b2 = (const float*)d_in[4];   // [512]
  float* out = (float*)d_out;                // [14400][512]

  char* ws = (char*)d_ws;
  size_t off = 0;
  unsigned short* xb  = (unsigned short*)(ws + off); off = alignup(off + (size_t)MROWS * D_MODEL * 2);
  unsigned short* w1t = (unsigned short*)(ws + off); off = alignup(off + (size_t)HD * D_MODEL * 2);
  unsigned short* w2t = (unsigned short*)(ws + off); off = alignup(off + (size_t)D_MODEL * HDP * 2);
  float*          h   = (float*)(ws + off);          off = alignup(off + (size_t)MROWS * HD * 4);
  float*          img = (float*)(ws + off);          off = alignup(off + (size_t)B2N * CCH * HH * WW * 4);
  unsigned short* h2  = (unsigned short*)(ws + off);

  // 1) x -> bf16
  {
    size_t n = (size_t)MROWS * D_MODEL;
    k_f32_to_bf16<<<(unsigned)((n + 255) / 256), 256, 0, stream>>>(x, xb, n);
  }
  // 2) W1 -> [1960][512] bf16 ; W2 -> [512][1984] bf16 (zero-padded K)
  {
    size_t n1 = (size_t)HD * D_MODEL;
    k_transpose_to_bf16<<<(unsigned)((n1 + 255) / 256), 256, 0, stream>>>(W1, w1t, D_MODEL, HD, D_MODEL);
    size_t n2 = (size_t)D_MODEL * HDP;
    k_transpose_to_bf16<<<(unsigned)((n2 + 255) / 256), 256, 0, stream>>>(W2, w2t, HD, D_MODEL, HDP);
  }
  // 3) GEMM1: h = x @ W1 + b1   (M=14400, N=1960, K=512)
  {
    int gm = (MROWS + TILE_M - 1) / TILE_M;     // 113
    int gn = (HD + TILE_N - 1) / TILE_N;        // 16
    k_gemm_bf16<<<gm * gn, 256, 0, stream>>>(xb, w1t, b1, h, MROWS, HD, D_MODEL);
  }
  // 4) fold + normalize
  {
    int n = B2N * CCH * HH * WW;
    k_fold_norm<<<(n + 255) / 256, 256, 0, stream>>>(h, img);
  }
  // 5) unfold + relu -> bf16 (with K pad columns zeroed)
  {
    size_t n = (size_t)MROWS * HDP;
    k_unfold_relu<<<(unsigned)((n + 255) / 256), 256, 0, stream>>>(img, h2);
  }
  // 6) GEMM2: out = relu(h') @ W2 + b2  (M=14400, N=512, K=1984)
  {
    int gm = (MROWS + TILE_M - 1) / TILE_M;     // 113
    int gn = (D_MODEL + TILE_N - 1) / TILE_N;   // 4
    k_gemm_bf16<<<gm * gn, 256, 0, stream>>>(h2, w2t, b2, out, MROWS, D_MODEL, HDP);
  }
}
